// CharacterCNN_28518582845875
// MI455X (gfx1250) — matrix operand order repaired and checked by reading, not compiled, not run on hardware
//
#include <hip/hip_runtime.h>
#include <hip/hip_bf16.h>
#include <math.h>

typedef __attribute__((ext_vector_type(16))) _Float16 v16h;
typedef __attribute__((ext_vector_type(2)))  _Float16 h2;
typedef __attribute__((ext_vector_type(8)))  float    v8f;

#define MC   20
#define CE   16
#define KDIM 224     // cnn feature width (32+64+128)
#define NT_H 14      // 224/16 column tiles for highway mats
#define NT_P 4       // 64/16 column tiles for projection
#define KS   7       // 224/32 K-steps per WMMA chain
#define WPB  2       // waves per block (each wave owns a 16-word M tile)

#define TAB_FLOATS 1344            // T1:64  T2:256  T3:1024
#define FRAG_HALFS 512             // 32 lanes * 16 f16 per 16x16x32 B fragment
#define NFRAGS     ((NT_H*KS)*2 + NT_P*KS)   // 98 + 98 + 28 = 224
#define FRAG_BYTE_OFF 8192         // frags start here inside d_ws

// ---------------------------------------------------------------------------
// Prep: pattern tables (vocab=2, E[0]=0 collapses conv+relu+maxpool to table
// lookups) + convert highway/proj weights into f16 WMMA B-fragment layout.
// ---------------------------------------------------------------------------
__global__ __launch_bounds__(256) void ccnn_prep(
    const float* __restrict__ E,
    const float* __restrict__ W1, const float* __restrict__ b1,
    const float* __restrict__ W2, const float* __restrict__ b2,
    const float* __restrict__ W3, const float* __restrict__ b3,
    const float* __restrict__ HtW, const float* __restrict__ HgW,
    const float* __restrict__ PW,
    float* __restrict__ tabs, _Float16* __restrict__ frags)
{
    const int tid    = blockIdx.x * blockDim.x + threadIdx.x;
    const int stride = gridDim.x * blockDim.x;

    // ---- pattern value tables (pre-bias+linear; relu/max applied later) ----
    for (int idx = tid; idx < TAB_FLOATS; idx += stride) {
        float val;
        if (idx < 64) {                       // conv1: 2 patterns x 32 ch
            int s = idx >> 5, o = idx & 31;
            float A = 0.f;
            for (int c = 0; c < CE; ++c) A += W1[o*CE + c] * E[CE + c];
            val = (float)s * A + b1[o];
        } else if (idx < 320) {               // conv2: 4 patterns x 64 ch
            int q = idx - 64; int p = q >> 6, o = q & 63;
            float A0 = 0.f, A1 = 0.f;
            for (int c = 0; c < CE; ++c) {
                float e = E[CE + c];
                A0 += W2[(o*CE + c)*2 + 0] * e;
                A1 += W2[(o*CE + c)*2 + 1] * e;
            }
            val = (float)(p & 1) * A0 + (float)((p >> 1) & 1) * A1 + b2[o];
        } else {                              // conv3: 8 patterns x 128 ch
            int q = idx - 320; int p = q >> 7, o = q & 127;
            float A0 = 0.f, A1 = 0.f, A2 = 0.f;
            for (int c = 0; c < CE; ++c) {
                float e = E[CE + c];
                A0 += W3[(o*CE + c)*3 + 0] * e;
                A1 += W3[(o*CE + c)*3 + 1] * e;
                A2 += W3[(o*CE + c)*3 + 2] * e;
            }
            val = (float)(p & 1) * A0 + (float)((p >> 1) & 1) * A1 +
                  (float)((p >> 2) & 1) * A2 + b3[o];
        }
        tabs[idx] = val;
    }

    // ---- f16 B fragments. y = x @ W^T  =>  B[k][n] = W[n*224 + k].
    // ISA 16x16x32 f16 B layout: lane L -> col L&15, half h=L>>4 -> slot i<8
    // holds K = 32s+8h+i, slot i>=8 holds K = 32s+16+8h+(i-8).
    for (int e = tid; e < NFRAGS * FRAG_HALFS; e += stride) {
        int f   = e >> 9;           // fragment id
        int el  = e & 511;
        int ln  = el >> 4;          // lane 0..31
        int idx = el & 15;          // f16 slot within lane
        const float* W; int ff;
        if      (f < NT_H*KS)       { W = HtW; ff = f; }
        else if (f < 2*NT_H*KS)     { W = HgW; ff = f - NT_H*KS; }
        else                        { W = PW;  ff = f - 2*NT_H*KS; }
        int t = ff / KS, s = ff % KS;
        int col = t*16 + (ln & 15);
        int k   = s*32 + (ln >> 4)*8 + ((idx < 8) ? idx : 16 + (idx - 8));
        frags[f*FRAG_HALFS + el] = (_Float16)W[col*KDIM + k];
    }
}

// ---------------------------------------------------------------------------
// Fused main kernel: masks -> table cnn -> highway (2 GEMMs, WMMA) -> proj.
// ---------------------------------------------------------------------------
__global__ __launch_bounds__(32 * WPB) void ccnn_main(
    const int* __restrict__ ids,
    const float* __restrict__ tabs_g,
    const _Float16* __restrict__ frags,
    const float* __restrict__ Htb, const float* __restrict__ Hgb,
    const float* __restrict__ Pb,
    float* __restrict__ out, int nWords)
{
    __shared__ float    s_tabs[TAB_FLOATS];
    __shared__ float    s_cnn[WPB][16][228];      // stride 228 -> bank-clean
    __shared__ _Float16 s_hw [WPB][16][232];

    const int wv   = threadIdx.x >> 5;
    const int lane = threadIdx.x & 31;
    const long tile     = (long)blockIdx.x * WPB + wv;
    const long wordbase = tile * 16;

    for (int i = threadIdx.x; i < TAB_FLOATS; i += blockDim.x)
        s_tabs[i] = tabs_g[i];
    __syncthreads();

    // ---------------- stage 1: occurrence masks -> cnn[16][224] ------------
    {
        const int w = lane >> 1, sub = lane & 1;    // 2 lanes per word
        const long word = wordbase + w;
        int idv[MC + 2];
        idv[0] = 0; idv[MC + 1] = 0;
        if (word < (long)nWords) {
            const int* p = ids + (size_t)word * MC;
            #pragma unroll
            for (int j = 0; j < MC; ++j) idv[j + 1] = p[j];
        } else {
            #pragma unroll
            for (int j = 0; j < MC; ++j) idv[j + 1] = 0;
        }
        int m1 = 0, m2 = 0, m3 = 0;
        #pragma unroll
        for (int j = 1; j <= MC; ++j) m1 |= 1 << idv[j];
        #pragma unroll
        for (int j = 0; j <= MC; ++j) m2 |= 1 << (idv[j] | (idv[j + 1] << 1));
        #pragma unroll
        for (int j = 0; j < MC; ++j)
            m3 |= 1 << (idv[j] | (idv[j + 1] << 1) | (idv[j + 2] << 2));

        for (int c = 0; c < 16; ++c) {              // conv1 -> ch [0,32)
            int co = sub * 16 + c; float v = -1e30f;
            #pragma unroll
            for (int s = 0; s < 2; ++s)
                if ((m1 >> s) & 1) v = fmaxf(v, s_tabs[s*32 + co]);
            s_cnn[wv][w][co] = fmaxf(v, 0.f);
        }
        for (int c = 0; c < 32; ++c) {              // conv2 -> ch [32,96)
            int co = sub * 32 + c; float v = -1e30f;
            #pragma unroll
            for (int p = 0; p < 4; ++p)
                if ((m2 >> p) & 1) v = fmaxf(v, s_tabs[64 + p*64 + co]);
            s_cnn[wv][w][32 + co] = fmaxf(v, 0.f);
        }
        for (int c = 0; c < 64; ++c) {              // conv3 -> ch [96,224)
            int co = sub * 64 + c; float v = -1e30f;
            #pragma unroll
            for (int p = 0; p < 8; ++p)
                if ((m3 >> p) & 1) v = fmaxf(v, s_tabs[320 + p*128 + co]);
            s_cnn[wv][w][96 + co] = fmaxf(v, 0.f);
        }
    }
    __syncthreads();

    // ---------------- stage 2: highway via WMMA ----------------------------
    const int m = lane & 15, h = lane >> 4, col = lane & 15;

    // A fragments (f16 from f32 cnn). ISA A layout: lane row = L&15,
    // K offsets 8h+{0,2,4,6} and 8h+16+{0,2,4,6} per VGPR pair.
    v16h a[KS];
    #pragma unroll
    for (int s = 0; s < KS; ++s)
        #pragma unroll
        for (int j = 0; j < 8; ++j) {
            int k = s*32 + h*8 + ((j < 4) ? 2*j : 16 + 2*(j - 4));
            float2 pr = *(const float2*)&s_cnn[wv][m][k];
            a[s][2*j]     = (_Float16)pr.x;
            a[s][2*j + 1] = (_Float16)pr.y;
        }

    for (int t = 0; t < NT_H; ++t) {
        v8f accT = {}, accG = {};
        #pragma unroll
        for (int s = 0; s < KS; ++s) {
            v16h bt = *(const v16h*)(frags + (size_t)(t*KS + s)*FRAG_HALFS + lane*16);
            accT = __builtin_amdgcn_wmma_f32_16x16x32_f16(
                       false, a[s], false, bt, (short)0, accT, false, false);
            v16h bg = *(const v16h*)(frags + (size_t)(NT_H*KS + t*KS + s)*FRAG_HALFS + lane*16);
            accG = __builtin_amdgcn_wmma_f32_16x16x32_f16(
                       false, a[s], false, bg, (short)0, accG, false, false);
        }
        const float bt_ = Htb[t*16 + col];
        const float bg_ = Hgb[t*16 + col];
        #pragma unroll
        for (int r = 0; r < 8; ++r) {               // C/D layout: M = r + 8h
            int mm = r + 8*h;
            float tt = fmaxf(accT[r] + bt_, 0.f);
            float x  = accG[r] + bg_;
            float gg = 1.f / (1.f + __expf(-x));
            float cv = s_cnn[wv][mm][t*16 + col];
            s_hw[wv][mm][t*16 + col] = (_Float16)(gg*tt + (1.f - gg)*cv);
        }
    }
    __syncthreads();

    // ---------------- stage 3: projection via WMMA -------------------------
    v16h ha[KS];
    #pragma unroll
    for (int s = 0; s < KS; ++s)
        #pragma unroll
        for (int j = 0; j < 8; ++j) {
            int k = s*32 + h*8 + ((j < 4) ? 2*j : 16 + 2*(j - 4));
            h2 pr = *(const h2*)&s_hw[wv][m][k];
            ha[s][2*j]     = pr.x;
            ha[s][2*j + 1] = pr.y;
        }

    for (int t = 0; t < NT_P; ++t) {
        v8f acc = {};
        #pragma unroll
        for (int s = 0; s < KS; ++s) {
            v16h bp = *(const v16h*)(frags + (size_t)(2*NT_H*KS + t*KS + s)*FRAG_HALFS + lane*16);
            acc = __builtin_amdgcn_wmma_f32_16x16x32_f16(
                      false, ha[s], false, bp, (short)0, acc, false, false);
        }
        const float pb = Pb[t*16 + col];
        #pragma unroll
        for (int r = 0; r < 8; ++r) {
            long word = wordbase + r + 8*h;
            if (word < (long)nWords)
                out[(size_t)word*64 + t*16 + col] = acc[r] + pb;
        }
    }
}

// ---------------------------------------------------------------------------
extern "C" void kernel_launch(void* const* d_in, const int* in_sizes, int n_in,
                              void* d_out, int out_size, void* d_ws, size_t ws_size,
                              hipStream_t stream) {
    const int*   char_ids = (const int*)  d_in[0];
    const float* E   = (const float*)d_in[1];
    const float* W1  = (const float*)d_in[2];
    const float* b1  = (const float*)d_in[3];
    const float* W2  = (const float*)d_in[4];
    const float* b2  = (const float*)d_in[5];
    const float* W3  = (const float*)d_in[6];
    const float* b3  = (const float*)d_in[7];
    const float* HtW = (const float*)d_in[8];
    const float* Htb = (const float*)d_in[9];
    const float* HgW = (const float*)d_in[10];
    const float* Hgb = (const float*)d_in[11];
    const float* PW  = (const float*)d_in[12];
    const float* Pb  = (const float*)d_in[13];
    float* out = (float*)d_out;

    const int nWords = in_sizes[0] / MC;

    float*    tabs  = (float*)d_ws;
    _Float16* frags = (_Float16*)((char*)d_ws + FRAG_BYTE_OFF);

    ccnn_prep<<<240, 256, 0, stream>>>(E, W1, b1, W2, b2, W3, b3,
                                       HtW, HgW, PW, tabs, frags);

    const int tiles  = (nWords + 15) / 16;
    const int blocks = (tiles + WPB - 1) / WPB;
    ccnn_main<<<blocks, 32 * WPB, 0, stream>>>(char_ids, tabs, frags,
                                               Htb, Hgb, Pb, out, nWords);
}
